// HypDistanceLinearLayer_64579128263384
// MI455X (gfx1250) — matrix operand order repaired and checked by reading, not compiled, not run on hardware
//
#include <hip/hip_runtime.h>
#include <hip/hip_bf16.h>
#include <math.h>

typedef __attribute__((ext_vector_type(16))) __bf16 v16bf;
typedef __attribute__((ext_vector_type(8)))  __bf16 v8bf;
typedef __attribute__((ext_vector_type(8)))  float  v8f;

#define BATCH   8192
#define OUT_DIM 2048
#define IN_DIM  512
#define EPSF      1e-5f
#define PROJ_EPSF 1e-3f

// ---------------------------------------------------------------------------
// Row preprocessing: one wave32 per row of 512 f32 (read-once -> NT loads).
// scale_mode=1: hyperbolic weight prep  w_ball = min(tanh(n),1-1e-3)/n * u
// scale_mode=0: plain bf16 cast of x
// Emits bf16 row (re-read by GEMM -> regular temporal) + squared norm.
// ---------------------------------------------------------------------------
template <int SCALE_MODE>
__global__ __launch_bounds__(256) void prep_rows(const float* __restrict__ src,
                                                 __bf16* __restrict__ dst_bf,
                                                 float* __restrict__ row_sq) {
    const int lane = threadIdx.x & 31;
    const int wave = threadIdx.x >> 5;
    const int row  = blockIdx.x * 8 + wave;
    const float* s = src + (size_t)row * IN_DIM;

    float v[16];
    float ss = 0.f;
#pragma unroll
    for (int i = 0; i < 16; ++i) {
        v[i] = __builtin_nontemporal_load(s + lane + 32 * i);
        ss += v[i] * v[i];
    }
#pragma unroll
    for (int off = 16; off; off >>= 1) ss += __shfl_xor(ss, off, 32);

    float scale = 1.0f;
    if (SCALE_MODE) {
        float nrm = sqrtf(ss);
        float cn  = fmaxf(nrm, EPSF);               // expmap0 norm clamp
        float tn  = tanhf(cn);                      // norm after expmap0
        float tgt = fminf(tn, 1.0f - PROJ_EPSF);    // ball projection
        scale = tgt / cn;
    }

    __bf16* d = dst_bf + (size_t)row * IN_DIM;
#pragma unroll
    for (int i = 0; i < 16; ++i) d[lane + 32 * i] = (__bf16)(v[i] * scale);

    if (lane == 0) row_sq[row] = ss * scale * scale;
}

// ---------------------------------------------------------------------------
// Fused WMMA GEMM (xy = x @ w^T, bf16 in / f32 acc) + hyperbolic distance
// epilogue. Block = 256 thr = 8 waves; block tile 128(M) x 128(N);
// wave tile 64(M) x 32(N) = 4x2 WMMA 16x16 tiles; K loop 512 in steps of 32.
// Fragment layouts used here (contiguous K, same order for A and B):
//   A (16x32 bf16): lane m(0-15)/m+16 holds row M=m, 16 contiguous K
//                   starting at k0 + ((lane>=16)?16:0)
//   B (32x16 bf16): lane n holds col N=n&15, 16 contiguous K starting at
//                   k0 + ((lane>=16)?16:0)   -> one 32B contiguous load
//   C/D f32: VGPR r -> M = r + 8*(lane>=16), N = lane&15
// Output is streamed with non-temporal stores (write-once, 64 MB).
// ---------------------------------------------------------------------------
__global__ __launch_bounds__(256) void hyp_gemm(const __bf16* __restrict__ xb,
                                                const __bf16* __restrict__ wb,
                                                const float* __restrict__ x2,
                                                const float* __restrict__ y2,
                                                float* __restrict__ out) {
    const int lane  = threadIdx.x & 31;
    const int wave  = threadIdx.x >> 5;
    const int wm    = wave >> 2;           // 0..1  -> 64-row strip
    const int wn    = wave & 3;            // 0..3  -> 32-col strip
    const int l15   = lane & 15;
    const int lhalf = lane >> 4;

    const int m0 = blockIdx.y * 128 + wm * 64;
    const int n0 = blockIdx.x * 128 + wn * 32;

    v8f acc[4][2];
#pragma unroll
    for (int mt = 0; mt < 4; ++mt)
#pragma unroll
        for (int nt = 0; nt < 2; ++nt) acc[mt][nt] = (v8f){};

    // per-lane fragment base pointers
    const __bf16* arow[4];
#pragma unroll
    for (int mt = 0; mt < 4; ++mt)
        arow[mt] = xb + (size_t)(m0 + mt * 16 + l15) * IN_DIM + (lhalf << 4);
    const __bf16* brow[2];
#pragma unroll
    for (int nt = 0; nt < 2; ++nt)
        brow[nt] = wb + (size_t)(n0 + nt * 16 + l15) * IN_DIM + (lhalf << 4);

#pragma unroll
    for (int k0 = 0; k0 < IN_DIM; k0 += 32) {
        v16bf a[4];
#pragma unroll
        for (int mt = 0; mt < 4; ++mt) {
            union { v16bf v; v8bf h[2]; } u;
            u.h[0] = *(const v8bf*)(arow[mt] + k0);        // K = base..base+7
            u.h[1] = *(const v8bf*)(arow[mt] + k0 + 8);    // K = base+8..+15
            a[mt] = u.v;
        }
        v16bf b[2];
#pragma unroll
        for (int nt = 0; nt < 2; ++nt)
            b[nt] = *(const v16bf*)(brow[nt] + k0);        // 32B contiguous

#pragma unroll
        for (int mt = 0; mt < 4; ++mt)
#pragma unroll
            for (int nt = 0; nt < 2; ++nt)
                acc[mt][nt] = __builtin_amdgcn_wmma_f32_16x16x32_bf16(
                    false, a[mt], false, b[nt], (short)0, acc[mt][nt],
                    false, false);
    }

    // ---- fused hyperbolic distance epilogue (c = 1) ----
    // d = -2*artanh(z) = ln((1-z)/(1+z)); z = sqrt(num2)/den >= 0 since
    // den = 1 - 2xy + x2*y2 + eps > 0 here, so only the upper clamp is live.
    float yv[2], yp[2];
#pragma unroll
    for (int nt = 0; nt < 2; ++nt) {
        yv[nt] = y2[n0 + nt * 16 + l15];
        yp[nt] = 1.f + yv[nt];
    }

#pragma unroll
    for (int mt = 0; mt < 4; ++mt) {
#pragma unroll
        for (int r = 0; r < 8; ++r) {
            const int   m   = m0 + mt * 16 + r + (lhalf << 3);
            const float xv  = x2[m];
            const float Bc  = 1.f - xv;
            const float Bc2 = Bc * Bc;
#pragma unroll
            for (int nt = 0; nt < 2; ++nt) {
                const float t    = acc[mt][nt][r] * 2.f;          // 2*xy
                const float A    = yp[nt] - t;
                const float den  = fmaf(xv, yv[nt], (1.f + EPSF) - t);
                const float num2 = fmaf(xv, A * A,
                                        fmaf(-t, A * Bc, yv[nt] * Bc2));
                const float s    = sqrtf(fmaxf(num2, 1e-12f));
                const float z    = fminf(__fdividef(s, den), 1.f - EPSF);
                const float res  = __logf(__fdividef(1.f - z, 1.f + z));
                const int   n    = n0 + nt * 16 + l15;
                __builtin_nontemporal_store(res, out + (size_t)m * OUT_DIM + n);
            }
        }
    }
}

// ---------------------------------------------------------------------------
extern "C" void kernel_launch(void* const* d_in, const int* in_sizes, int n_in,
                              void* d_out, int out_size, void* d_ws, size_t ws_size,
                              hipStream_t stream) {
    const float* x = (const float*)d_in[0];      // [8192, 512]
    const float* w = (const float*)d_in[1];      // [2048, 512]
    float* out = (float*)d_out;                  // [8192, 2048]

    // workspace: xb bf16 (8MB) | wb bf16 (2MB) | x2 f32 (32KB) | y2 f32 (8KB)
    __bf16* xb = (__bf16*)d_ws;
    __bf16* wb = xb + (size_t)BATCH * IN_DIM;
    float*  x2 = (float*)(wb + (size_t)OUT_DIM * IN_DIM);
    float*  y2 = x2 + BATCH;

    prep_rows<0><<<BATCH / 8, 256, 0, stream>>>(x, xb, x2);
    prep_rows<1><<<OUT_DIM / 8, 256, 0, stream>>>(w, wb, y2);

    dim3 grid(OUT_DIM / 128, BATCH / 128);   // 16 x 64
    hyp_gemm<<<grid, 256, 0, stream>>>(xb, wb, x2, y2, out);
}
